// Attention_68023692034760
// MI455X (gfx1250) — matrix operand order repaired and checked by reading, not compiled, not run on hardware
//
#include <hip/hip_runtime.h>
#include <stdint.h>
#include <stddef.h>

// ---------------------------------------------------------------------------
// Types for CDNA5 WMMA (wave32): v_wmma_f32_16x16x32_bf16
// ---------------------------------------------------------------------------
typedef __attribute__((ext_vector_type(16))) __bf16 v16bf;
typedef __attribute__((ext_vector_type(8)))  __bf16 v8bf;
typedef __attribute__((ext_vector_type(8)))  float  v8f;
typedef __attribute__((ext_vector_type(4)))  unsigned int v4u;
typedef __attribute__((ext_vector_type(8)))  int v8i_;
typedef __attribute__((ext_vector_type(4)))  int v4i_;

#define B_     2
#define N_     2048
#define H_     8
#define D_     64
#define DIM_   512
#define QKVC_  1536
#define KCONV_ 33

// ---------------------------------------------------------------------------
// Wave helpers (wave32; avoid __AMDGCN_WAVEFRONT_SIZE-dependent HIP shuffles)
// ---------------------------------------------------------------------------
__device__ __forceinline__ int lane_id() {
  return (int)__builtin_amdgcn_mbcnt_lo(~0u, 0u);
}

__device__ __forceinline__ float xor_bcast(float v, int m) {
  int src = (lane_id() ^ m) << 2;
  int r = __builtin_amdgcn_ds_bpermute(src, __builtin_bit_cast(int, v));
  return __builtin_bit_cast(float, r);
}

__device__ __forceinline__ v8f zero8() {
  v8f z;
#pragma unroll
  for (int i = 0; i < 8; ++i) z[i] = 0.0f;
  return z;
}

__device__ __forceinline__ v8f wmma_bf16(v16bf a, v16bf b, v8f c) {
  return __builtin_amdgcn_wmma_f32_16x16x32_bf16(false, a, false, b,
                                                 (short)0, c, false, false);
}

// ---------------------------------------------------------------------------
// Compiler-visibility shim: TDM / async asm write LDS via raw offsets, which
// LLVM cannot see as stores to the __shared__ object.  Escape the LDS pointer
// into an opaque asm with a memory clobber so subsequent ds_loads must read
// the real (DMA-written) contents instead of folding to undef.
// ---------------------------------------------------------------------------
__device__ __forceinline__ void lds_publish(void* p) {
  asm volatile("" :: "v"(p) : "memory");
}

// ---------------------------------------------------------------------------
// Async copy global->LDS (ASYNCcnt path), 16 bytes per lane per issue.
// ---------------------------------------------------------------------------
__device__ __forceinline__ void async_copy_b128(uint32_t lds_off, uint64_t gaddr) {
  asm volatile("global_load_async_to_lds_b128 %0, %1, off"
               :: "v"(lds_off), "v"(gaddr) : "memory");
}
__device__ __forceinline__ void wait_asynccnt0() {
  asm volatile("s_wait_asynccnt 0x0" ::: "memory");
}

// ---------------------------------------------------------------------------
// Tensor Data Mover: 2D bf16 tile load, global -> (padded) LDS.
// D# per CDNA5 ISA 8.3/8.4: group0 {count=1, lds_addr, global_addr, type=2},
// group1 {data_size=2B, pad cfg, tensor dims, tile dims, dim0 stride}.
// pad_interval code 3 = 16 DWORDs (64B row), pad_amount code 3 = 4 DWORDs.
// ---------------------------------------------------------------------------
__device__ __forceinline__ void tdm_load_2d(
    uint32_t lds_off, const void* gptr,
    uint32_t tensor_d0, uint32_t tensor_d1,
    uint32_t tile_d0, uint32_t tile_d1,
    uint32_t stride0, uint32_t pad_interval, uint32_t pad_amount) {
  uint64_t ga = (uint64_t)(uintptr_t)gptr;
  v4u g0;
  g0[0] = 1u;                                   // count=1, user descriptor
  g0[1] = lds_off;                              // [63:32] lds_addr
  g0[2] = (uint32_t)(ga & 0xffffffffu);         // [95:64] global_addr lo
  g0[3] = (uint32_t)((ga >> 32) & 0x1ffffffu)   // [120:96] global_addr hi
        | (2u << 30);                           // [127:126] type=2 (image)
  v8i_ g1;
  uint32_t flags = (1u << 16)                   // data_size = 2 bytes
                 | (1u << 20)                   // pad_enable
                 | (pad_interval << 22) | (pad_amount << 25);
  g1[0] = (int)flags;
  g1[1] = (int)((tensor_d0 & 0xffffu) << 16);                     // abar=0 | td0.lo
  g1[2] = (int)((tensor_d0 >> 16) | ((tensor_d1 & 0xffffu) << 16));
  g1[3] = (int)((tensor_d1 >> 16) | (tile_d0 << 16));
  g1[4] = (int)tile_d1;                                           // tile_d2=0
  g1[5] = (int)stride0;                                           // dim0 stride lo
  g1[6] = 0;                                                      // stride hi / s1 lo
  g1[7] = 0;
  v4i_ z4 = {0, 0, 0, 0};
#if __clang_major__ >= 23
  v8i_ z8 = {0, 0, 0, 0, 0, 0, 0, 0};
  __builtin_amdgcn_tensor_load_to_lds(g0, g1, z4, z4, z8, 0);
#else
  __builtin_amdgcn_tensor_load_to_lds(g0, g1, z4, z4, 0);
#endif
}

// ---------------------------------------------------------------------------
// Fragment loaders matching the ISA 16-bit A (16x32) / B (32x16) VGPR layouts.
// ---------------------------------------------------------------------------
__device__ __forceinline__ v16bf load_frag_a(const __bf16* p, int ld) {
  const int lane = threadIdx.x & 31;
  const int row  = lane & 15;
  const int khi  = (lane >> 4) << 3;
  const __bf16* q = p + row * ld + khi;
  v8bf lo = *(const v8bf*)(q);
  v8bf hi = *(const v8bf*)(q + 16);
  v16bf f;
#pragma unroll
  for (int e = 0; e < 8; ++e) { f[e] = lo[e]; f[e + 8] = hi[e]; }
  return f;
}

__device__ __forceinline__ v16bf load_frag_b(const __bf16* p, int ld) {
  const int lane = threadIdx.x & 31;
  const int col  = lane & 15;
  const int kb   = (lane >> 4) << 3;
  const __bf16* q = p + col * ld + kb;
  v8bf lo = *(const v8bf*)(q);
  v8bf hi = *(const v8bf*)(q + 16);
  v16bf f;
#pragma unroll
  for (int e = 0; e < 8; ++e) { f[e] = lo[e]; f[e + 8] = hi[e]; }
  return f;
}

// ---------------------------------------------------------------------------
// K0: fp32 -> bf16 conversions; W_qkv / W_out transposed to (col, k) layout
// ---------------------------------------------------------------------------
__global__ __launch_bounds__(256) void k_convert(
    const float* __restrict__ x, const float* __restrict__ wqkv,
    const float* __restrict__ wout, __bf16* __restrict__ xb,
    __bf16* __restrict__ wqkvT, __bf16* __restrict__ woutT) {
  const int stride = gridDim.x * blockDim.x;
  const int tid = blockIdx.x * blockDim.x + threadIdx.x;
  const int nx = B_ * N_ * DIM_;
  for (int i = tid; i < nx; i += stride) xb[i] = (__bf16)x[i];
  const int nq = QKVC_ * DIM_;
  for (int i = tid; i < nq; i += stride) {
    int c = i >> 9, k = i & 511;
    wqkvT[i] = (__bf16)wqkv[k * QKVC_ + c];
  }
  const int nw = DIM_ * DIM_;
  for (int i = tid; i < nw; i += stride) {
    int c = i >> 9, k = i & 511;
    woutT[i] = (__bf16)wout[k * DIM_ + c];
  }
}

// ---------------------------------------------------------------------------
// K1: qkv = x @ W_qkv  (4096x512 @ 512x1536), TDM double-buffered staging.
// LDS: A0@0, B0@5120, A1@10240, B1@15360; tiles [64][40] bf16 (TDM-padded).
// ---------------------------------------------------------------------------
__global__ __launch_bounds__(128) void k_qkv_gemm(
    const __bf16* __restrict__ xb, const __bf16* __restrict__ wqkvT,
    __bf16* __restrict__ qb, __bf16* __restrict__ kb,
    __bf16* __restrict__ vb, float* __restrict__ vf) {
  __shared__ __align__(16) char smem[4 * 5120];
  const int rowBase = blockIdx.y * 64;
  const int colBase = blockIdx.x * 64;
  const int t = threadIdx.x;
  const int wave = t >> 5, lane = t & 31;
  const int wr = (wave >> 1) * 32, wc = (wave & 1) * 32;

  v8f acc[2][2];
#pragma unroll
  for (int i = 0; i < 2; ++i)
#pragma unroll
    for (int j = 0; j < 2; ++j) acc[i][j] = zero8();

  if (wave == 0) {
    tdm_load_2d(0u,    xb    + (size_t)rowBase * DIM_, DIM_, B_ * N_, 32, 64, DIM_, 3, 3);
    tdm_load_2d(5120u, wqkvT + (size_t)colBase * DIM_, DIM_, QKVC_,   32, 64, DIM_, 3, 3);
  }

  const int NIT = DIM_ / 32;
  for (int it = 0; it < NIT; ++it) {
    __syncthreads();  // compute(it-1) done -> buf[(it+1)&1] free for DMA
    if (wave == 0) {
      if (it + 1 < NIT) {
        uint32_t bo = (uint32_t)(((it + 1) & 1) * 10240);
        tdm_load_2d(bo,        xb    + (size_t)rowBase * DIM_ + (it + 1) * 32,
                    DIM_, B_ * N_, 32, 64, DIM_, 3, 3);
        tdm_load_2d(bo + 5120, wqkvT + (size_t)colBase * DIM_ + (it + 1) * 32,
                    DIM_, QKVC_, 32, 64, DIM_, 3, 3);
        __builtin_amdgcn_s_wait_tensorcnt(2);  // buf[it&1] complete
      } else {
        __builtin_amdgcn_s_wait_tensorcnt(0);
      }
    }
    __syncthreads();      // publish buf[it&1]
    lds_publish(smem);    // make DMA-written LDS visible to the compiler
    const __bf16* As = (const __bf16*)(smem + (it & 1) * 10240);
    const __bf16* Bs = (const __bf16*)(smem + (it & 1) * 10240 + 5120);
    v16bf fa0 = load_frag_a(As + wr * 40, 40);
    v16bf fa1 = load_frag_a(As + (wr + 16) * 40, 40);
    v16bf fb0 = load_frag_b(Bs + wc * 40, 40);
    v16bf fb1 = load_frag_b(Bs + (wc + 16) * 40, 40);
    acc[0][0] = wmma_bf16(fa0, fb0, acc[0][0]);
    acc[0][1] = wmma_bf16(fa0, fb1, acc[0][1]);
    acc[1][0] = wmma_bf16(fa1, fb0, acc[1][0]);
    acc[1][1] = wmma_bf16(fa1, fb1, acc[1][1]);
  }

  const int halfoff = (lane >> 4) << 3;
  const int colLane = lane & 15;
#pragma unroll
  for (int mi = 0; mi < 2; ++mi)
#pragma unroll
    for (int ni = 0; ni < 2; ++ni)
#pragma unroll
      for (int r = 0; r < 8; ++r) {
        int grow = rowBase + wr + mi * 16 + r + halfoff;
        int gcol = colBase + wc + ni * 16 + colLane;
        float v = acc[mi][ni][r];
        int chunk = gcol >> 9, rem = gcol & 511;
        int hh = rem >> 6, dd = rem & 63;
        int bidx = grow >> 11, nn = grow & 2047;
        size_t di = ((size_t)(bidx * H_ + hh) * N_ + nn) * D_ + dd;
        if (chunk == 0)      qb[di] = (__bf16)(v * 0.125f);  // fold 1/sqrt(d)
        else if (chunk == 1) kb[di] = (__bf16)v;
        else { vb[di] = (__bf16)v; vf[di] = v; }
      }
}

// ---------------------------------------------------------------------------
// K2: flash attention; Q/K tiles staged via global_load_async_to_lds_b128.
// LDS: Qs@0, Ks@9216, Vs@18432 (transposed [d][key]), Ps@27648; ld = 72 bf16.
// ---------------------------------------------------------------------------
__global__ __launch_bounds__(128) void k_attn(
    const __bf16* __restrict__ qb, const __bf16* __restrict__ kb,
    const __bf16* __restrict__ vb, const unsigned char* __restrict__ mask,
    float* __restrict__ aout) {
  __shared__ __align__(16) char smem[4 * 9216];
  __bf16* Qs = (__bf16*)(smem);
  __bf16* Ks = (__bf16*)(smem + 9216);
  __bf16* Vs = (__bf16*)(smem + 18432);
  __bf16* Ps = (__bf16*)(smem + 27648);

  const int bh = blockIdx.y;
  const int bidx = bh >> 3, hh = bh & 7;
  const int rowBase = blockIdx.x * 64;
  const int t = threadIdx.x, wave = t >> 5, lane = t & 31;
  const int halfoff = (lane >> 4) << 3;
  const int colLane = lane & 15;

  const __bf16* Qg = qb + ((size_t)(bidx * H_ + hh) * N_ + rowBase) * D_;
  const __bf16* Kg = kb + ((size_t)(bidx * H_ + hh) * N_) * D_;
  const __bf16* Vg = vb + ((size_t)(bidx * H_ + hh) * N_) * D_;
  const unsigned char* mrow = mask + bidx * N_;

  const int sr = t >> 1, sc = (t & 1) << 5;  // staging row / col (32 elems)

  {  // stage Q tile 64x64 via async copy
    uint64_t g = (uint64_t)(uintptr_t)(Qg + (size_t)sr * D_ + sc);
    uint32_t l = (uint32_t)sr * 144u + (uint32_t)sc * 2u;
#pragma unroll
    for (int e = 0; e < 4; ++e) async_copy_b128(l + 16u * e, g + 16u * e);
  }

  v8f oacc[4];
#pragma unroll
  for (int i = 0; i < 4; ++i) oacc[i] = zero8();
  float mrun[8], lrun[8];
#pragma unroll
  for (int r = 0; r < 8; ++r) { mrun[r] = -3.0e38f; lrun[r] = 0.0f; }

  for (int j = 0; j < N_; j += 64) {
    __syncthreads();  // previous tile fully consumed
    {  // stage K tile via async copy; V transposed via ds stores
      uint64_t g = (uint64_t)(uintptr_t)(Kg + (size_t)(j + sr) * D_ + sc);
      uint32_t l = 9216u + (uint32_t)sr * 144u + (uint32_t)sc * 2u;
#pragma unroll
      for (int e = 0; e < 4; ++e) async_copy_b128(l + 16u * e, g + 16u * e);
      const __bf16* gv = Vg + (size_t)(j + sr) * D_ + sc;
#pragma unroll
      for (int e = 0; e < 32; ++e) Vs[(sc + e) * 72 + sr] = gv[e];
    }
    wait_asynccnt0();
    __syncthreads();
    lds_publish(smem);  // async-written Q/K tiles now compiler-visible

    // S = Q K^T  (this wave: 16 rows x 64 keys)
    v8f sacc[4];
#pragma unroll
    for (int i = 0; i < 4; ++i) sacc[i] = zero8();
#pragma unroll
    for (int kd = 0; kd < 64; kd += 32) {
      v16bf aQ = load_frag_a(Qs + (wave * 16) * 72 + kd, 72);
#pragma unroll
      for (int ni = 0; ni < 4; ++ni) {
        v16bf bK = load_frag_b(Ks + (ni * 16) * 72 + kd, 72);
        sacc[ni] = wmma_bf16(aQ, bK, sacc[ni]);
      }
    }

    // key mask (per output column)
#pragma unroll
    for (int ni = 0; ni < 4; ++ni) {
      bool kp = mrow[j + ni * 16 + colLane] != 0;
#pragma unroll
      for (int r = 0; r < 8; ++r)
        sacc[ni][r] = kp ? sacc[ni][r] : -3.0e38f;
    }

    // online softmax
    float alpha[8];
#pragma unroll
    for (int r = 0; r < 8; ++r) {
      float m = fmaxf(fmaxf(sacc[0][r], sacc[1][r]),
                      fmaxf(sacc[2][r], sacc[3][r]));
#pragma unroll
      for (int sh = 1; sh < 16; sh <<= 1) m = fmaxf(m, xor_bcast(m, sh));
      float mnew = fmaxf(mrun[r], m);
      alpha[r] = __expf(mrun[r] - mnew);
      mrun[r] = mnew;
    }
    float rsum[8];
#pragma unroll
    for (int r = 0; r < 8; ++r) rsum[r] = 0.0f;
#pragma unroll
    for (int ni = 0; ni < 4; ++ni)
#pragma unroll
      for (int r = 0; r < 8; ++r) {
        float p = __expf(sacc[ni][r] - mrun[r]);
        sacc[ni][r] = p;
        rsum[r] += p;
      }
#pragma unroll
    for (int r = 0; r < 8; ++r) {
      float s = rsum[r];
#pragma unroll
      for (int sh = 1; sh < 16; sh <<= 1) s += xor_bcast(s, sh);
      lrun[r] = lrun[r] * alpha[r] + s;
    }
#pragma unroll
    for (int nd = 0; nd < 4; ++nd)
#pragma unroll
      for (int r = 0; r < 8; ++r) oacc[nd][r] *= alpha[r];

    // C-layout -> A-layout relayout of P through wave-private LDS strip
#pragma unroll
    for (int ni = 0; ni < 4; ++ni)
#pragma unroll
      for (int r = 0; r < 8; ++r)
        Ps[(wave * 16 + r + halfoff) * 72 + ni * 16 + colLane] =
            (__bf16)sacc[ni][r];
    asm volatile("s_wait_dscnt 0" ::: "memory");  // wave-local LDS RAW

    // O += P V
#pragma unroll
    for (int kd = 0; kd < 64; kd += 32) {
      v16bf aP = load_frag_a(Ps + (wave * 16) * 72 + kd, 72);
#pragma unroll
      for (int nd = 0; nd < 4; ++nd) {
        v16bf bV = load_frag_b(Vs + (nd * 16) * 72 + kd, 72);
        oacc[nd] = wmma_bf16(aP, bV, oacc[nd]);
      }
    }
  }

  // normalize and write (b, n, h*64+d) fp32
#pragma unroll
  for (int r = 0; r < 8; ++r) lrun[r] = 1.0f / lrun[r];
#pragma unroll
  for (int nd = 0; nd < 4; ++nd)
#pragma unroll
    for (int r = 0; r < 8; ++r) {
      int row = rowBase + wave * 16 + r + halfoff;
      int col = hh * D_ + nd * 16 + colLane;
      aout[((size_t)bidx * N_ + row) * DIM_ + col] = oacc[nd][r] * lrun[r];
    }
}

// ---------------------------------------------------------------------------
// K3: depthwise conv residual over sequence axis + add; emit bf16 GEMM operand
// ---------------------------------------------------------------------------
__global__ __launch_bounds__(256) void k_conv_add(
    const float* __restrict__ vf, const float* __restrict__ aout,
    const float* __restrict__ convw, __bf16* __restrict__ zb) {
  __shared__ float cw[H_ * KCONV_];
  for (int i = threadIdx.x; i < H_ * KCONV_; i += blockDim.x) cw[i] = convw[i];
  __syncthreads();
  const int total = B_ * H_ * N_ * D_;
  for (int idx = blockIdx.x * blockDim.x + threadIdx.x; idx < total;
       idx += gridDim.x * blockDim.x) {
    int d = idx & 63;
    int i = (idx >> 6) & (N_ - 1);
    int h = (idx >> 17) & 7;
    int b = idx >> 20;
    const float* vp = vf + ((size_t)(b * H_ + h) * N_) * D_ + d;
    float res = 0.0f;
#pragma unroll
    for (int tt = 0; tt < KCONV_; ++tt) {
      int ii = i + tt - (KCONV_ / 2);
      if (ii >= 0 && ii < N_) res += cw[h * KCONV_ + tt] * vp[(size_t)ii * D_];
    }
    size_t oi = ((size_t)b * N_ + i) * DIM_ + h * D_ + d;
    zb[oi] = (__bf16)(aout[oi] + res);
  }
}

// ---------------------------------------------------------------------------
// K4: out = z @ W_out + b_out, TDM double-buffered staging, mask epilogue
// ---------------------------------------------------------------------------
__global__ __launch_bounds__(128) void k_out_gemm(
    const __bf16* __restrict__ zb, const __bf16* __restrict__ woutT,
    const float* __restrict__ bout, const unsigned char* __restrict__ mask,
    float* __restrict__ out) {
  __shared__ __align__(16) char smem[4 * 5120];
  const int rowBase = blockIdx.y * 64;
  const int colBase = blockIdx.x * 64;
  const int t = threadIdx.x;
  const int wave = t >> 5, lane = t & 31;
  const int wr = (wave >> 1) * 32, wc = (wave & 1) * 32;

  v8f acc[2][2];
#pragma unroll
  for (int i = 0; i < 2; ++i)
#pragma unroll
    for (int j = 0; j < 2; ++j) acc[i][j] = zero8();

  if (wave == 0) {
    tdm_load_2d(0u,    zb    + (size_t)rowBase * DIM_, DIM_, B_ * N_, 32, 64, DIM_, 3, 3);
    tdm_load_2d(5120u, woutT + (size_t)colBase * DIM_, DIM_, DIM_,    32, 64, DIM_, 3, 3);
  }

  const int NIT = DIM_ / 32;
  for (int it = 0; it < NIT; ++it) {
    __syncthreads();
    if (wave == 0) {
      if (it + 1 < NIT) {
        uint32_t bo = (uint32_t)(((it + 1) & 1) * 10240);
        tdm_load_2d(bo,        zb    + (size_t)rowBase * DIM_ + (it + 1) * 32,
                    DIM_, B_ * N_, 32, 64, DIM_, 3, 3);
        tdm_load_2d(bo + 5120, woutT + (size_t)colBase * DIM_ + (it + 1) * 32,
                    DIM_, DIM_, 32, 64, DIM_, 3, 3);
        __builtin_amdgcn_s_wait_tensorcnt(2);
      } else {
        __builtin_amdgcn_s_wait_tensorcnt(0);
      }
    }
    __syncthreads();
    lds_publish(smem);
    const __bf16* As = (const __bf16*)(smem + (it & 1) * 10240);
    const __bf16* Bs = (const __bf16*)(smem + (it & 1) * 10240 + 5120);
    v16bf fa0 = load_frag_a(As + wr * 40, 40);
    v16bf fa1 = load_frag_a(As + (wr + 16) * 40, 40);
    v16bf fb0 = load_frag_b(Bs + wc * 40, 40);
    v16bf fb1 = load_frag_b(Bs + (wc + 16) * 40, 40);
    acc[0][0] = wmma_bf16(fa0, fb0, acc[0][0]);
    acc[0][1] = wmma_bf16(fa0, fb1, acc[0][1]);
    acc[1][0] = wmma_bf16(fa1, fb0, acc[1][0]);
    acc[1][1] = wmma_bf16(fa1, fb1, acc[1][1]);
  }

  const int halfoff = (lane >> 4) << 3;
  const int colLane = lane & 15;
#pragma unroll
  for (int mi = 0; mi < 2; ++mi)
#pragma unroll
    for (int ni = 0; ni < 2; ++ni)
#pragma unroll
      for (int r = 0; r < 8; ++r) {
        int grow = rowBase + wr + mi * 16 + r + halfoff;
        int gcol = colBase + wc + ni * 16 + colLane;
        float v = acc[mi][ni][r] + bout[gcol];
        out[(size_t)grow * DIM_ + gcol] = (mask[grow] != 0) ? v : 0.0f;
      }
}

// ---------------------------------------------------------------------------
// Launcher: fixed workspace layout (~38 MB)
// ---------------------------------------------------------------------------
extern "C" void kernel_launch(void* const* d_in, const int* in_sizes, int n_in,
                              void* d_out, int out_size, void* d_ws,
                              size_t ws_size, hipStream_t stream) {
  const float* x = (const float*)d_in[0];
  const unsigned char* mask = (const unsigned char*)d_in[1];  // jnp bool = 1B
  const float* wqkv = (const float*)d_in[2];
  const float* wout = (const float*)d_in[3];
  const float* bout = (const float*)d_in[4];
  const float* convw = (const float*)d_in[5];
  float* out = (float*)d_out;

  char* ws = (char*)d_ws;
  __bf16* xb    = (__bf16*)(ws + 0);          // 4096x512 bf16   (4.0 MB)
  __bf16* wqkvT = (__bf16*)(ws + 4194304);    // 1536x512 bf16   (1.5 MB)
  __bf16* woutT = (__bf16*)(ws + 5767168);    // 512x512 bf16    (0.5 MB)
  __bf16* qb    = (__bf16*)(ws + 6291456);    // (b,h,n,d) bf16  (4.0 MB)
  __bf16* kb    = (__bf16*)(ws + 10485760);   // (b,h,n,d) bf16  (4.0 MB)
  __bf16* vb    = (__bf16*)(ws + 14680064);   // (b,h,n,d) bf16  (4.0 MB)
  float*  vf    = (float*) (ws + 18874368);   // (b,h,n,d) f32   (8.0 MB)
  float*  aoutf = (float*) (ws + 27262976);   // (b,n,512) f32   (8.0 MB)
  __bf16* zb    = (__bf16*)(ws + 35651584);   // (b*n,512) bf16  (4.0 MB)

  k_convert<<<2048, 256, 0, stream>>>(x, wqkv, wout, xb, wqkvT, woutT);
  k_qkv_gemm<<<dim3(QKVC_ / 64, (B_ * N_) / 64), 128, 0, stream>>>(
      xb, wqkvT, qb, kb, vb, vf);
  k_attn<<<dim3(N_ / 64, B_ * H_), 128, 0, stream>>>(qb, kb, vb, mask, aoutf);
  k_conv_add<<<1024, 256, 0, stream>>>(vf, aoutf, convw, zb);
  k_out_gemm<<<dim3(DIM_ / 64, (B_ * N_) / 64), 128, 0, stream>>>(
      zb, woutT, bout, mask, out);
}
